// VoxelBackBoneHiRes_89773406421479
// MI455X (gfx1250) — matrix operand order repaired and checked by reading, not compiled, not run on hardware
//
#include <hip/hip_runtime.h>
#include <hip/hip_bf16.h>

// MI455X / gfx1250 implementation of VoxelBackBoneHiRes:
//   2x (submanifold 3x3x3 sparse conv as gather->WMMA GEMM, K=432) + BN + ReLU
//   + 1x1 branch conv + segment-sum merge + final [U,16]x[16,128] WMMA GEMM.
// Matrix math: v_wmma_f32_16x16x32_bf16 (f32 accumulate).
// Gather inputs are staged into zero-padded buffers (row N == 0) so missing
// neighbors cost a single address select -- no mask multiplies, no fused-BN
// math in the WMMA inner loop.  BN+ReLU between the convs is one cheap
// elementwise pass (bandwidth-trivial vs. the 27x gather traffic).

typedef __attribute__((ext_vector_type(16))) __bf16 v16bf;
typedef __attribute__((ext_vector_type(8)))  float  v8f;

#define BN_EPS 1e-3f
#define NPAIR  14   // ceil(27/2) neighbor-offset pairs (K = 28*16, last half zero)

__device__ __forceinline__ unsigned pack_bf2(float lo, float hi) {
  union { __bf16 h[2]; unsigned u; } x;
  x.h[0] = (__bf16)lo; x.h[1] = (__bf16)hi;
  return x.u;
}

// ---------------------------------------------------------------------------
// zero fill
// ---------------------------------------------------------------------------
__global__ void vbb_zero_f32(float* __restrict__ p, long long n) {
  long long i = (long long)blockIdx.x * blockDim.x + threadIdx.x;
  if (i < n) p[i] = 0.0f;
}

// ---------------------------------------------------------------------------
// copy x -> xp with a zero pad row at index N  (xp has (N+1)*16 elements)
// ---------------------------------------------------------------------------
__global__ void vbb_pad_copy(const float* __restrict__ x, float* __restrict__ xp,
                             long long n16) {          // n16 = (N+1)*16
  long long i = (long long)blockIdx.x * blockDim.x + threadIdx.x;
  if (i < n16) xp[i] = (i < n16 - 16) ? x[i] : 0.0f;
}

// ---------------------------------------------------------------------------
// h = relu(y*a + c) with a zero pad row at index N (h has (N+1)*16 elements)
// ---------------------------------------------------------------------------
__global__ void vbb_bn_relu_pad(const float* __restrict__ y,
                                const float* __restrict__ bn,   // [32] a,c
                                float* __restrict__ h, long long n16) {
  long long i = (long long)blockIdx.x * blockDim.x + threadIdx.x;
  if (i >= n16) return;
  const int c = (int)(i & 15);
  h[i] = (i < n16 - 16) ? fmaxf(y[i] * bn[c] + bn[16 + c], 0.0f) : 0.0f;
}

// ---------------------------------------------------------------------------
// Submanifold conv (gather -> WMMA).  Input xp is zero-padded: row N == 0,
// so a missing neighbor (-1) just selects row N.  One wave per FULL 16-row
// tile; K-loop fully unrolled (14 WMMAs); branchless stores.
// ---------------------------------------------------------------------------
__global__ void vbb_subm_conv_wmma(const float* __restrict__ xp,   // [N+1,16]
                                   const float* __restrict__ W,    // [27,16,16]
                                   const int*   __restrict__ nbr,  // [N,27]
                                   float* __restrict__ y,          // [N,16]
                                   float* __restrict__ stats,      // [32] sum,sumsq
                                   int ntiles, int N)
{
  __shared__ unsigned wfrag[NPAIR * 32 * 8]; // prebuilt B fragments (bf16 pairs)
  __shared__ float sred[16], qred[16];

  const int tid  = threadIdx.x;
  const int lane = tid & 31;
  const int wave = tid >> 5;
  const int tile = blockIdx.x * (blockDim.x >> 5) + wave;
  const bool live = (tile < ntiles);         // wave-uniform

  const int m = lane & 15;
  const int g = lane >> 4;                   // selects neighbor offset within the pair
  const int row = (live ? tile : 0) * 16 + m;
  const int* nrow = nbr + (long long)row * 27;
  __builtin_prefetch(nrow, 0, 0);            // global_prefetch of index row

  if (tid < 16) { sred[tid] = 0.0f; qred[tid] = 0.0f; }

  for (int i = tid; i < NPAIR * 32 * 8; i += blockDim.x) {
    const int p  = i >> 8;                   // pair
    const int ln = (i >> 3) & 31;
    const int v  = i & 7;                    // dword within lane (2 K-values)
    const int k  = 2 * p + (ln >> 4);
    const int n  = ln & 15;
    unsigned u = 0u;
    if (k < 27)
      u = pack_bf2(W[k * 256 + (2 * v) * 16 + n],
                   W[k * 256 + (2 * v + 1) * 16 + n]);
    wfrag[i] = u;
  }
  __syncthreads();

  float s_acc = 0.0f, q_acc = 0.0f;

  if (live) {                                // wave-uniform: EXEC all-1s inside
    v8f acc = {};

#pragma unroll
    for (int p = 0; p < NPAIR; ++p) {
      const int i0r = nrow[2 * p];
      const int i1r = (2 * p + 1 < 27) ? nrow[2 * p + 1] : -1;
      const int i0  = (i0r < 0) ? N : i0r;   // zero pad row
      const int i1  = (i1r < 0) ? N : i1r;
      const float4* p0 = (const float4*)(xp + (size_t)(g ? i1 : i0) * 16);
      const float4* p1 = (const float4*)(xp + (size_t)(g ? i1 : i0) * 16 + 8);
      const float4 u0 = p0[0], u1 = p0[1];
      const float4 w0 = p1[0], w1 = p1[1];

      v16bf a;
      a[0]  = (__bf16)u0.x; a[1]  = (__bf16)u0.y; a[2]  = (__bf16)u0.z; a[3]  = (__bf16)u0.w;
      a[4]  = (__bf16)u1.x; a[5]  = (__bf16)u1.y; a[6]  = (__bf16)u1.z; a[7]  = (__bf16)u1.w;
      a[8]  = (__bf16)w0.x; a[9]  = (__bf16)w0.y; a[10] = (__bf16)w0.z; a[11] = (__bf16)w0.w;
      a[12] = (__bf16)w1.x; a[13] = (__bf16)w1.y; a[14] = (__bf16)w1.z; a[15] = (__bf16)w1.w;

      const v16bf b = *(const v16bf*)&wfrag[(p * 32 + lane) * 8];
      acc = __builtin_amdgcn_wmma_f32_16x16x32_bf16(false, a, false, b,
                                                    (short)0, acc, false, false);
    }

    // D layout: VGPR j holds row (j | j+8) of the tile, col = lane&15.
    const int nb    = lane & 15;
    const int mbase = tile * 16 + ((lane < 16) ? 0 : 8);
    float* py = y + (size_t)mbase * 16 + nb;  // one addr, offset-encoded stores
#pragma unroll
    for (int j = 0; j < 8; ++j) {
      const float d = acc[j];
      py[j * 16] = d;
      s_acc += d;  q_acc += d * d;
    }
  }

  // block-level BN-stat reduction, then 32 global atomics per block
  atomicAdd(&sred[lane & 15], s_acc);
  atomicAdd(&qred[lane & 15], q_acc);
  __syncthreads();
  if (tid < 16) {
    atomicAdd(&stats[tid],      sred[tid]);
    atomicAdd(&stats[16 + tid], qred[tid]);
  }
}

// ---------------------------------------------------------------------------
// Scalar tail for the conv (rows base..N-1, < 16 rows): one thread per
// (row, cout).  Gathers from the padded buffer; folds into BN stats.
// ---------------------------------------------------------------------------
__global__ void vbb_subm_conv_tail(const float* __restrict__ xp,
                                   const float* __restrict__ W,
                                   const int*   __restrict__ nbr,
                                   float* __restrict__ y,
                                   float* __restrict__ stats,
                                   int base, int N)
{
  const int t = blockIdx.x * blockDim.x + threadIdx.x;
  const int row = base + (t >> 4);
  const int c   = t & 15;
  if (row >= N) return;
  float acc = 0.0f;
  for (int k = 0; k < 27; ++k) {
    const int ir  = nbr[(long long)row * 27 + k];
    const int idx = (ir < 0) ? N : ir;
    const float* xr = xp + (size_t)idx * 16;
    const float* wk = W + k * 256 + c;
#pragma unroll
    for (int ci = 0; ci < 16; ++ci)
      acc += (float)(__bf16)xr[ci] * (float)(__bf16)wk[ci * 16];
  }
  y[(size_t)row * 16 + c] = acc;
  atomicAdd(&stats[c], acc);
  atomicAdd(&stats[16 + c], acc * acc);
}

// ---------------------------------------------------------------------------
// BN finalize: a = gamma * rsqrt(var+eps), c = beta - mu*a
// ---------------------------------------------------------------------------
__global__ void vbb_bn_finalize(const float* __restrict__ stats,
                                const float* __restrict__ gamma,
                                const float* __restrict__ beta,
                                float* __restrict__ bn, float invN)
{
  const int t = threadIdx.x;
  if (t < 16) {
    const float mu  = stats[t] * invN;
    const float var = stats[16 + t] * invN - mu * mu;
    const float a   = __frsqrt_rn(var + BN_EPS) * gamma[t];
    bn[t]      = a;
    bn[16 + t] = beta[t] - mu * a;
  }
}

// ---------------------------------------------------------------------------
// merge: rows 0..N-1  -> relu(bn1(y1))       -> merged[seg[r]]
//        rows N..2N-1 -> feat2 @ Wobo + bobo -> merged[seg[r]]
// ---------------------------------------------------------------------------
__global__ void vbb_merge(const float* __restrict__ y1,
                          const float* __restrict__ bn1,
                          const float* __restrict__ feat2,
                          const float* __restrict__ Wobo,   // [16,16]
                          const float* __restrict__ bobo,   // [16]
                          const int*   __restrict__ seg,    // [2N]
                          float* __restrict__ merged, int N)
{
  __shared__ float sW[256], sb[16], sbn[32];
  const int tid = threadIdx.x;
  if (tid < 256) sW[tid] = Wobo[tid];
  if (tid < 16)  sb[tid] = bobo[tid];
  if (tid < 32)  sbn[tid] = bn1[tid];
  __syncthreads();

  const long long t = (long long)blockIdx.x * blockDim.x + tid;
  const long long total = 2LL * N * 16;
  if (t >= total) return;
  const int       c = (int)(t & 15);
  const long long r = t >> 4;
  float v;
  if (r < N) {
    v = fmaxf(y1[r * 16 + c] * sbn[c] + sbn[16 + c], 0.0f);
  } else {
    const long long r2 = r - N;
    float acc = sb[c];
#pragma unroll
    for (int k = 0; k < 16; ++k) acc += feat2[r2 * 16 + k] * sW[k * 16 + c];
    v = acc;
  }
  atomicAdd(&merged[(long long)seg[r] * 16 + c], v);
}

// ---------------------------------------------------------------------------
// final GEMM (full tiles only): out[U,128] = merged[U,16] @ Wf[16,128] + bias
// K=16 zero-padded to 32 inside the bf16 WMMA fragments.
// ---------------------------------------------------------------------------
__global__ void vbb_final_wmma(const float* __restrict__ merged,
                               const float* __restrict__ Wf,    // [16,128]
                               const float* __restrict__ bias,  // [128]
                               float* __restrict__ out, int utiles)
{
  __shared__ unsigned wfrag[8 * 32 * 8];   // 8 N-tiles of Wf
  const int tid = threadIdx.x;
  for (int i = tid; i < 8 * 32 * 8; i += blockDim.x) {
    const int nt = i >> 8;
    const int ln = (i >> 3) & 31;
    const int v  = i & 7;
    unsigned u = 0u;
    if (ln < 16) {                          // lanes>=16 carry K=16..31 -> zero pad
      const int n = ln & 15;
      u = pack_bf2(Wf[(2 * v) * 128 + nt * 16 + n],
                   Wf[(2 * v + 1) * 128 + nt * 16 + n]);
    }
    wfrag[i] = u;
  }
  __syncthreads();

  const int lane = tid & 31;
  const int wave = tid >> 5;
  const int tile = blockIdx.x * (blockDim.x >> 5) + wave;
  if (tile >= utiles) return;               // wave-uniform

  const int m = lane & 15;
  const int g = lane >> 4;
  const float* src = merged + (size_t)(tile * 16 + m) * 16;

  v16bf a;
#pragma unroll
  for (int j = 0; j < 8; ++j) { a[j] = (__bf16)(g ? 0.0f : src[j]); a[8 + j] = (__bf16)(g ? 0.0f : src[8 + j]); }

  const int nb    = lane & 15;
  const int mbase = tile * 16 + ((lane < 16) ? 0 : 8);

#pragma unroll
  for (int nt = 0; nt < 8; ++nt) {
    v8f acc = {};
    const v16bf b = *(const v16bf*)&wfrag[(nt * 32 + lane) * 8];
    acc = __builtin_amdgcn_wmma_f32_16x16x32_bf16(false, a, false, b,
                                                  (short)0, acc, false, false);
    const float bval = bias[nt * 16 + nb];
    float* po = out + (size_t)mbase * 128 + nt * 16 + nb;
#pragma unroll
    for (int j = 0; j < 8; ++j)
      po[j * 128] = acc[j] + bval;          // offset-encoded branchless stores
  }
}

// ---------------------------------------------------------------------------
// Scalar tail for the final GEMM (rows base..U-1, < 16 rows).
// ---------------------------------------------------------------------------
__global__ void vbb_final_tail(const float* __restrict__ merged,
                               const float* __restrict__ Wf,
                               const float* __restrict__ bias,
                               float* __restrict__ out, int base, int U)
{
  const int t = blockIdx.x * blockDim.x + threadIdx.x;
  const int row = base + (t >> 7);
  const int c   = t & 127;
  if (row >= U) return;
  float acc = bias[c];
  const float* mr = merged + (size_t)row * 16;
#pragma unroll
  for (int k = 0; k < 16; ++k)
    acc += (float)(__bf16)mr[k] * (float)(__bf16)Wf[k * 128 + c];
  out[(size_t)row * 128 + c] = acc;
}

// ---------------------------------------------------------------------------
extern "C" void kernel_launch(void* const* d_in, const int* in_sizes, int n_in,
                              void* d_out, int out_size, void* d_ws, size_t ws_size,
                              hipStream_t stream)
{
  const float* x     = (const float*)d_in[0];
  const float* feat2 = (const float*)d_in[1];
  const float* W0    = (const float*)d_in[2];
  const float* g0    = (const float*)d_in[3];
  const float* b0    = (const float*)d_in[4];
  const float* W1    = (const float*)d_in[5];
  const float* g1    = (const float*)d_in[6];
  const float* b1    = (const float*)d_in[7];
  const float* Wobo  = (const float*)d_in[8];
  const float* bobo  = (const float*)d_in[9];
  const float* Wf    = (const float*)d_in[10];
  const float* bfv   = (const float*)d_in[11];
  const int*   nbr   = (const int*)d_in[12];
  const int*   seg   = (const int*)d_in[13];
  (void)n_in; (void)ws_size;

  const int N = in_sizes[0] / 16;
  const int U = out_size / 128;
  float* out = (float*)d_out;

  // workspace carve-up
  float* ws     = (float*)d_ws;
  float* xp     = ws;                          // (N+1)*16  padded input
  float* h0     = xp + (size_t)(N + 1) * 16;   // (N+1)*16  padded relu(bn(y0))
  float* y0     = h0 + (size_t)(N + 1) * 16;   // N*16
  float* y1     = y0 + (size_t)N * 16;         // N*16
  float* stats0 = y1 + (size_t)N * 16;         // 32
  float* bn0    = stats0 + 32;                 // 32
  float* stats1 = bn0 + 32;                    // 32
  float* bn1    = stats1 + 32;                 // 32
  float* merged = bn1 + 32;                    // U*16

  const long long np16 = (long long)(N + 1) * 16;
  const long long nz   = (long long)U * 16;
  vbb_zero_f32<<<(unsigned)((nz + 255) / 256), 256, 0, stream>>>(merged, nz);
  vbb_zero_f32<<<1, 128, 0, stream>>>(stats0, 128); // stats0,bn0,stats1,bn1
  vbb_pad_copy<<<(unsigned)((np16 + 255) / 256), 256, 0, stream>>>(x, xp, np16);

  const int ntiles = N >> 4;                  // full tiles only
  const int ntail  = N & 15;
  const int wpb    = 8;                       // 8 wave32 per 256-thread block
  const int blocks = (ntiles + wpb - 1) / wpb;

  vbb_subm_conv_wmma<<<blocks, 256, 0, stream>>>(xp, W0, nbr, y0, stats0, ntiles, N);
  if (ntail)
    vbb_subm_conv_tail<<<1, 256, 0, stream>>>(xp, W0, nbr, y0, stats0, ntiles * 16, N);
  vbb_bn_finalize<<<1, 32, 0, stream>>>(stats0, g0, b0, bn0, 1.0f / (float)N);

  vbb_bn_relu_pad<<<(unsigned)((np16 + 255) / 256), 256, 0, stream>>>(y0, bn0, h0, np16);

  vbb_subm_conv_wmma<<<blocks, 256, 0, stream>>>(h0, W1, nbr, y1, stats1, ntiles, N);
  if (ntail)
    vbb_subm_conv_tail<<<1, 256, 0, stream>>>(h0, W1, nbr, y1, stats1, ntiles * 16, N);
  vbb_bn_finalize<<<1, 32, 0, stream>>>(stats1, g1, b1, bn1, 1.0f / (float)N);

  const long long tm = 2LL * N * 16;
  vbb_merge<<<(unsigned)((tm + 255) / 256), 256, 0, stream>>>(y1, bn1, feat2, Wobo, bobo, seg, merged, N);

  const int utiles  = U >> 4;                 // full tiles only
  const int utail   = U & 15;
  const int ublocks = (utiles + wpb - 1) / wpb;
  if (utiles)
    vbb_final_wmma<<<ublocks, 256, 0, stream>>>(merged, Wf, bfv, out, utiles);
  if (utail)
    vbb_final_tail<<<(utail * 128 + 255) / 256, 256, 0, stream>>>(merged, Wf, bfv, out, utiles * 16, U);
}
